// TransformerBlock_25769804134
// MI455X (gfx1250) — matrix operand order repaired and checked by reading, not compiled, not run on hardware
//
#include <hip/hip_runtime.h>
#include <hip/hip_bf16.h>
#include <math.h>

#define DIM   768
#define HID   3072
#define HEADS 12
#define SEQ   1024
#define NTOK  8192      // 8 * 1024
#define EPS   1e-5f

typedef __attribute__((ext_vector_type(16))) __bf16 bf16x16;
typedef __attribute__((ext_vector_type(8)))  float  f32x8;

struct alignas(16) U4 { unsigned int x, y, z, w; };

union Frag16 { U4 q[2]; bf16x16 v; };

__device__ __forceinline__ f32x8 wmma_bf16f32(const Frag16& a, const Frag16& b, f32x8 c) {
  // D = A(16x32 bf16) x B(32x16 bf16) + C(16x16 f32)
  return __builtin_amdgcn_wmma_f32_16x16x32_bf16(false, a.v, false, b.v, (short)0, c,
                                                 false, false);
}

// CDNA5 async copy global -> LDS (16B), tracked by ASYNCcnt.
// LDS operand: low 32 bits of the flat-cast shared pointer == wave LDS offset.
__device__ __forceinline__ void async_b128(const void* g, void* l) {
  unsigned loff = (unsigned)(unsigned long long)l;
  asm volatile("global_load_async_to_lds_b128 %0, %1, off"
               :: "v"(loff), "v"(g) : "memory");
}
__device__ __forceinline__ void wait_async() {
  asm volatile("s_wait_asynccnt 0x0" ::: "memory");
}

__device__ __forceinline__ float halfmax16(float x) {
#pragma unroll
  for (int m = 1; m < 16; m <<= 1) x = fmaxf(x, __shfl_xor(x, m, 32));
  return x;
}
__device__ __forceinline__ float halfsum16(float x) {
#pragma unroll
  for (int m = 1; m < 16; m <<= 1) x += __shfl_xor(x, m, 32);
  return x;
}

// ---------------------------------------------------------------------------
// f32 -> bf16 elementwise cast (weights; done per launch, stays deterministic)
// ---------------------------------------------------------------------------
__global__ void cvt_kernel(const float* __restrict__ in, __hip_bfloat16* __restrict__ out, int n) {
  for (int i = blockIdx.x * blockDim.x + threadIdx.x; i < n; i += gridDim.x * blockDim.x)
    out[i] = __float2bfloat16(in[i]);
}

// ---------------------------------------------------------------------------
// LayerNorm over last dim (768) + cast to bf16. One block per token row.
// ---------------------------------------------------------------------------
__global__ __launch_bounds__(256) void ln_kernel(const float* __restrict__ x,
                                                 const float* __restrict__ g,
                                                 const float* __restrict__ bb,
                                                 __hip_bfloat16* __restrict__ out) {
  const int row = blockIdx.x;
  const float* xr = x + (size_t)row * DIM;
  float s = 0.f, s2 = 0.f;
  for (int i = threadIdx.x; i < DIM; i += 256) { float v = xr[i]; s += v; s2 += v * v; }
#pragma unroll
  for (int m = 1; m < 32; m <<= 1) { s += __shfl_xor(s, m, 32); s2 += __shfl_xor(s2, m, 32); }
  __shared__ float sh[16];
  const int wid = threadIdx.x >> 5;
  if ((threadIdx.x & 31) == 0) { sh[wid] = s; sh[wid + 8] = s2; }
  __syncthreads();
  s = sh[threadIdx.x & 7];
  s2 = sh[(threadIdx.x & 7) + 8];
#pragma unroll
  for (int m = 1; m < 8; m <<= 1) { s += __shfl_xor(s, m, 32); s2 += __shfl_xor(s2, m, 32); }
  const float mu  = s * (1.0f / DIM);
  const float var = s2 * (1.0f / DIM) - mu * mu;
  const float rs  = rsqrtf(var + EPS);
  __hip_bfloat16* orow = out + (size_t)row * DIM;
  for (int i = threadIdx.x; i < DIM; i += 256)
    orow[i] = __float2bfloat16((xr[i] - mu) * rs * g[i] + bb[i]);
}

// ---------------------------------------------------------------------------
// bf16 WMMA GEMM: C[M,N] = A[M,K] @ B[K,N] + bias, fused epilogue.
//   EPI 0: bias            -> bf16 out
//   EPI 1: bias + residual -> f32 out
//   EPI 2: bias + GELU     -> bf16 out
// Block tile 128x128x32, 256 threads = 8 waves, each wave 32x64 (2x4 frags).
// Tiles staged with GLOBAL_LOAD_ASYNC_TO_LDS_B128 (double-buffered).
// LDS row strides padded (20 / 68 dwords) for bank-conflict-free frag loads.
// ---------------------------------------------------------------------------
#define ASTR 20
#define BSTR 68

template <int EPI>
__global__ __launch_bounds__(256) void gemm_bf16_kernel(
    const __hip_bfloat16* __restrict__ A, const __hip_bfloat16* __restrict__ B,
    const float* __restrict__ bias, const float* __restrict__ resid,
    float* __restrict__ Cf, __hip_bfloat16* __restrict__ Cb,
    int M, int N, int K) {
  __shared__ alignas(16) unsigned int ldsA[2][128 * ASTR];
  __shared__ alignas(16) unsigned int ldsB[2][32 * BSTR];

  const int tid  = threadIdx.x;
  const int n0   = blockIdx.x * 128;
  const int m0   = blockIdx.y * 128;
  const int wid  = tid >> 5;
  const int lane = tid & 31;
  const int wm   = (wid & 3) * 32;   // wave row base within block tile
  const int wn   = (wid >> 2) * 64;  // wave col base within block tile
  const int hi   = lane >> 4;
  const int lm   = lane & 15;

  // per-thread staging coordinates
  const int ar0 = tid >> 2, aseg = tid & 3;      // A: rows 0..63 (+64), seg 16B
  const int br0 = tid >> 4, bseg = tid & 15;     // B: rows 0..15 (+16), seg 16B

  f32x8 acc[2][4];
#pragma unroll
  for (int i = 0; i < 2; i++)
#pragma unroll
    for (int j = 0; j < 4; j++) acc[i][j] = (f32x8)0.0f;

  // prologue: stage K-tile 0 via async copy
#pragma unroll
  for (int i = 0; i < 2; i++) {
    const int row = ar0 + i * 64;
    async_b128(A + (size_t)(m0 + row) * K + aseg * 8, &ldsA[0][row * ASTR + aseg * 4]);
  }
#pragma unroll
  for (int i = 0; i < 2; i++) {
    const int row = br0 + i * 16;
    async_b128(B + (size_t)row * N + n0 + bseg * 8, &ldsB[0][row * BSTR + bseg * 4]);
  }
  wait_async();
  __syncthreads();

  const int nk = K / 32;
  for (int ki = 0; ki < nk; ki++) {
    const int cur = ki & 1;
    if (ki + 1 < nk) {
      const int nxt = cur ^ 1;
      const int kt  = (ki + 1) * 32;
#pragma unroll
      for (int i = 0; i < 2; i++) {
        const int row = ar0 + i * 64;
        async_b128(A + (size_t)(m0 + row) * K + kt + aseg * 8,
                   &ldsA[nxt][row * ASTR + aseg * 4]);
      }
#pragma unroll
      for (int i = 0; i < 2; i++) {
        const int row = br0 + i * 16;
        async_b128(B + (size_t)(kt + row) * N + n0 + bseg * 8,
                   &ldsB[nxt][row * BSTR + bseg * 4]);
      }
    }
    // fragment loads (ISA 16-bit A/B layouts)
    Frag16 af[2], bfrag[4];
#pragma unroll
    for (int am = 0; am < 2; am++) {
      const int mrow = wm + am * 16 + lm;
      af[am].q[0] = *(const U4*)&ldsA[cur][mrow * ASTR + hi * 4];
      af[am].q[1] = *(const U4*)&ldsA[cur][mrow * ASTR + 8 + hi * 4];
    }
#pragma unroll
    for (int bn = 0; bn < 4; bn++) {
      const int ncol = wn + bn * 16 + lm;
      const __bf16* Bs = (const __bf16*)ldsB[cur];
#pragma unroll
      for (int e = 0; e < 8; e++) {
        bfrag[bn].v[e]     = Bs[(hi * 8 + e) * (2 * BSTR) + ncol];
        bfrag[bn].v[8 + e] = Bs[(16 + hi * 8 + e) * (2 * BSTR) + ncol];
      }
    }
#pragma unroll
    for (int am = 0; am < 2; am++)
#pragma unroll
      for (int bn = 0; bn < 4; bn++)
        acc[am][bn] = wmma_bf16f32(af[am], bfrag[bn], acc[am][bn]);
    wait_async();           // next-buffer fills landed (no-op on last iter)
    __syncthreads();
  }

  // epilogue (C layout: lane -> N, vgpr -> M, halves offset M by 8)
#pragma unroll
  for (int am = 0; am < 2; am++) {
#pragma unroll
    for (int bn = 0; bn < 4; bn++) {
      const int n   = n0 + wn + bn * 16 + lm;
      const float bia = bias[n];
      const int mb  = m0 + wm + am * 16 + hi * 8;
#pragma unroll
      for (int v = 0; v < 8; v++) {
        const size_t idx = (size_t)(mb + v) * N + n;
        float val = acc[am][bn][v] + bia;
        if (EPI == 0) {
          Cb[idx] = __float2bfloat16(val);
        } else if (EPI == 1) {
          Cf[idx] = val + resid[idx];
        } else {
          const float ge = 0.5f * val * (1.0f + erff(val * 0.70710678118654752f));
          Cb[idx] = __float2bfloat16(ge);
        }
      }
    }
  }
}

// ---------------------------------------------------------------------------
// Flash attention (non-causal). grid = (SEQ/128, B*HEADS), 256 threads.
// Block: one (b,h), 128 query rows. Each wave owns 16 rows.
// Q and V staged via async-to-LDS; K staged transposed (VGPR path) so the
// Q@K^T B-fragments are contiguous in LDS. P is re-shaped through LDS from
// C-layout to A-fragment layout for P@V.
// ---------------------------------------------------------------------------
__global__ __launch_bounds__(256) void attn_kernel(const __hip_bfloat16* __restrict__ qkv,
                                                   __hip_bfloat16* __restrict__ o) {
  __shared__ alignas(16) unsigned int ldsQ[128 * 36];
  __shared__ alignas(16) unsigned int ldsKt[64 * 36];
  __shared__ alignas(16) unsigned int ldsV[64 * 36];
  __shared__ alignas(16) unsigned int ldsP[128 * 36];

  const int tid = threadIdx.x, wid = tid >> 5, lane = tid & 31;
  const int hi = lane >> 4, lm = lane & 15;
  const int bh = blockIdx.y;
  const int b = bh / HEADS, h = bh % HEADS;
  const int q0 = blockIdx.x * 128;
  const size_t tokbase = (size_t)b * SEQ;
  const int QN = 3 * DIM;
  const float scale = 0.125f;   // 1/sqrt(64)

  // stage Q chunk [128 x 64] bf16 (async)
#pragma unroll
  for (int i = 0; i < 4; i++) {
    int c = tid + 256 * i;               // 1024 chunks of 16B
    int row = c >> 3, seg = c & 7;
    async_b128(qkv + (tokbase + q0 + row) * QN + h * 64 + seg * 8,
               &ldsQ[row * 36 + seg * 4]);
  }
  wait_async();
  __syncthreads();

  const int r0 = wid * 16;
  Frag16 aq[2];
#pragma unroll
  for (int s = 0; s < 2; s++) {
    const int mrow = r0 + lm;
    aq[s].q[0] = *(const U4*)&ldsQ[mrow * 36 + s * 16 + hi * 4];
    aq[s].q[1] = *(const U4*)&ldsQ[mrow * 36 + s * 16 + 8 + hi * 4];
  }

  f32x8 accO[4];
#pragma unroll
  for (int i = 0; i < 4; i++) accO[i] = (f32x8)0.0f;
  float rowm[8], rowl[8];
#pragma unroll
  for (int v = 0; v < 8; v++) { rowm[v] = -1e30f; rowl[v] = 0.0f; }

  __hip_bfloat16* Ktb = (__hip_bfloat16*)ldsKt;
  __hip_bfloat16* Pb  = (__hip_bfloat16*)ldsP;

  for (int kt = 0; kt < SEQ / 64; kt++) {
    // stage K^T [64d x 64j] (transposed scatter) and V [64j x 64d] (async)
    {
      const int j = tid >> 2, d0 = (tid & 3) * 16;
      const size_t ra = (tokbase + (size_t)kt * 64 + j) * QN + h * 64;
      U4 k0 = *(const U4*)(qkv + ra + DIM + d0);
      U4 k1 = *(const U4*)(qkv + ra + DIM + d0 + 8);
      const __hip_bfloat16* kp = (const __hip_bfloat16*)&k0;
#pragma unroll
      for (int e = 0; e < 8; e++) Ktb[(d0 + e) * 72 + j] = kp[e];
      kp = (const __hip_bfloat16*)&k1;
#pragma unroll
      for (int e = 0; e < 8; e++) Ktb[(d0 + 8 + e) * 72 + j] = kp[e];
      async_b128(qkv + ra + 2 * DIM + d0,     &ldsV[j * 36 + d0 / 2]);
      async_b128(qkv + ra + 2 * DIM + d0 + 8, &ldsV[j * 36 + d0 / 2 + 4]);
    }
    wait_async();
    __syncthreads();

    // S = Q @ K^T  (16 x 64 per wave)
    f32x8 sfr[4];
#pragma unroll
    for (int nf = 0; nf < 4; nf++) sfr[nf] = (f32x8)0.0f;
#pragma unroll
    for (int s = 0; s < 2; s++) {
#pragma unroll
      for (int nf = 0; nf < 4; nf++) {
        Frag16 bk;
#pragma unroll
        for (int e = 0; e < 8; e++) {
          bk.v[e]     = ((const __bf16*)Ktb)[(s * 32 + hi * 8 + e) * 72 + nf * 16 + lm];
          bk.v[8 + e] = ((const __bf16*)Ktb)[(s * 32 + 16 + hi * 8 + e) * 72 + nf * 16 + lm];
        }
        sfr[nf] = wmma_bf16f32(aq[s], bk, sfr[nf]);
      }
    }
#pragma unroll
    for (int nf = 0; nf < 4; nf++)
#pragma unroll
      for (int v = 0; v < 8; v++) sfr[nf][v] *= scale;

    // online softmax update (per-row stats live replicated in 16-lane halves)
#pragma unroll
    for (int v = 0; v < 8; v++) {
      float mx = sfr[0][v];
#pragma unroll
      for (int nf = 1; nf < 4; nf++) mx = fmaxf(mx, sfr[nf][v]);
      mx = halfmax16(mx);
      const float mn = fmaxf(rowm[v], mx);
      const float alpha = __expf(rowm[v] - mn);
      rowm[v] = mn;
      rowl[v] *= alpha;
#pragma unroll
      for (int nf = 0; nf < 4; nf++) accO[nf][v] *= alpha;
      float rs = 0.0f;
#pragma unroll
      for (int nf = 0; nf < 4; nf++) {
        sfr[nf][v] = __expf(sfr[nf][v] - mn);
        rs += sfr[nf][v];
      }
      rowl[v] += halfsum16(rs);
    }

    // spill P (bf16) to LDS to transpose into A-fragment layout
#pragma unroll
    for (int nf = 0; nf < 4; nf++)
#pragma unroll
      for (int v = 0; v < 8; v++)
        Pb[(r0 + v + hi * 8) * 72 + nf * 16 + lm] = __float2bfloat16(sfr[nf][v]);
    __syncthreads();

    // O += P @ V
#pragma unroll
    for (int s = 0; s < 2; s++) {
      Frag16 ap;
      const int mrow = r0 + lm;
      ap.q[0] = *(const U4*)&ldsP[mrow * 36 + s * 16 + hi * 4];
      ap.q[1] = *(const U4*)&ldsP[mrow * 36 + s * 16 + 8 + hi * 4];
#pragma unroll
      for (int nf = 0; nf < 4; nf++) {
        Frag16 bv;
#pragma unroll
        for (int e = 0; e < 8; e++) {
          bv.v[e]     = ((const __bf16*)ldsV)[(s * 32 + hi * 8 + e) * 72 + nf * 16 + lm];
          bv.v[8 + e] = ((const __bf16*)ldsV)[(s * 32 + 16 + hi * 8 + e) * 72 + nf * 16 + lm];
        }
        accO[nf] = wmma_bf16f32(ap, bv, accO[nf]);
      }
    }
    __syncthreads();
  }

  // normalize + store bf16 into [token][h*64+d]
#pragma unroll
  for (int nf = 0; nf < 4; nf++)
#pragma unroll
    for (int v = 0; v < 8; v++) {
      const int m = r0 + v + hi * 8;
      const float val = accO[nf][v] / rowl[v];
      o[(tokbase + q0 + m) * DIM + h * 64 + nf * 16 + lm] = __float2bfloat16(val);
    }
}

// ---------------------------------------------------------------------------
extern "C" void kernel_launch(void* const* d_in, const int* in_sizes, int n_in,
                              void* d_out, int out_size, void* d_ws, size_t ws_size,
                              hipStream_t stream) {
  const float* x      = (const float*)d_in[0];
  const float* ln1_g  = (const float*)d_in[1];
  const float* ln1_b  = (const float*)d_in[2];
  const float* qkv_w  = (const float*)d_in[3];
  const float* qkv_b  = (const float*)d_in[4];
  const float* proj_w = (const float*)d_in[5];
  const float* proj_b = (const float*)d_in[6];
  const float* ln2_g  = (const float*)d_in[7];
  const float* ln2_b  = (const float*)d_in[8];
  const float* fc1_w  = (const float*)d_in[9];
  const float* fc1_b  = (const float*)d_in[10];
  const float* fc2_w  = (const float*)d_in[11];
  const float* fc2_b  = (const float*)d_in[12];
  float* out = (float*)d_out;

  char* w = (char*)d_ws;
  __hip_bfloat16* qkvw_bf = (__hip_bfloat16*)w;  w += (size_t)DIM * 3 * DIM * 2;
  __hip_bfloat16* projw_bf = (__hip_bfloat16*)w; w += (size_t)DIM * DIM * 2;
  __hip_bfloat16* fc1w_bf = (__hip_bfloat16*)w;  w += (size_t)DIM * HID * 2;
  __hip_bfloat16* fc2w_bf = (__hip_bfloat16*)w;  w += (size_t)HID * DIM * 2;
  __hip_bfloat16* xn_bf   = (__hip_bfloat16*)w;  w += (size_t)NTOK * DIM * 2;
  __hip_bfloat16* qkv_bf  = (__hip_bfloat16*)w;  w += (size_t)NTOK * 3 * DIM * 2;
  __hip_bfloat16* o_bf    = (__hip_bfloat16*)w;  w += (size_t)NTOK * DIM * 2;
  float*          x1      = (float*)w;           w += (size_t)NTOK * DIM * 4;
  __hip_bfloat16* h_bf    = (__hip_bfloat16*)w;  w += (size_t)NTOK * HID * 2;

  // weight casts to bf16
  cvt_kernel<<<2048, 256, 0, stream>>>(qkv_w, qkvw_bf, DIM * 3 * DIM);
  cvt_kernel<<<2048, 256, 0, stream>>>(proj_w, projw_bf, DIM * DIM);
  cvt_kernel<<<2048, 256, 0, stream>>>(fc1_w, fc1w_bf, DIM * HID);
  cvt_kernel<<<2048, 256, 0, stream>>>(fc2_w, fc2w_bf, HID * DIM);

  // --- attention sublayer ---
  ln_kernel<<<NTOK, 256, 0, stream>>>(x, ln1_g, ln1_b, xn_bf);
  gemm_bf16_kernel<0><<<dim3(3 * DIM / 128, NTOK / 128), 256, 0, stream>>>(
      xn_bf, qkvw_bf, qkv_b, nullptr, nullptr, qkv_bf, NTOK, 3 * DIM, DIM);
  attn_kernel<<<dim3(SEQ / 128, (NTOK / SEQ) * HEADS), 256, 0, stream>>>(qkv_bf, o_bf);
  gemm_bf16_kernel<1><<<dim3(DIM / 128, NTOK / 128), 256, 0, stream>>>(
      o_bf, projw_bf, proj_b, x, x1, nullptr, NTOK, DIM, DIM);

  // --- MLP sublayer ---
  ln_kernel<<<NTOK, 256, 0, stream>>>(x1, ln2_g, ln2_b, xn_bf);
  gemm_bf16_kernel<2><<<dim3(HID / 128, NTOK / 128), 256, 0, stream>>>(
      xn_bf, fc1w_bf, fc1_b, nullptr, nullptr, h_bf, NTOK, HID, DIM);
  gemm_bf16_kernel<1><<<dim3(DIM / 128, NTOK / 128), 256, 0, stream>>>(
      h_bf, fc2w_bf, fc2_b, x1, out, nullptr, NTOK, DIM, HID);
}
